// focus_89283780149274
// MI455X (gfx1250) — matrix operand order repaired and checked by reading, not compiled, not run on hardware
//
#include <hip/hip_runtime.h>

#define DIMS 1024
#define HEADS 16
#define HD 64
#define NB 4
#define CTXN 1024
#define NROWS 4096            // NB * CTXN
#define NELEM 4194304         // NROWS * DIMS
#define NWIN 16               // CTXN / HD
#define D2 32                 // HD / 2
#define EPSF 1.1920929e-7f    // FLT_EPSILON (jnp.finfo(f32).eps)

typedef _Float16 f16;
typedef __attribute__((ext_vector_type(16))) _Float16 v16h;
typedef __attribute__((ext_vector_type(8)))  _Float16 v8h;
typedef __attribute__((ext_vector_type(8)))  float    v8f;
typedef __attribute__((ext_vector_type(4)))  int      v4i;

// ---- optional gfx1250 async global->LDS path (ASYNCcnt-tracked) ----
#if defined(__has_builtin)
#if __has_builtin(__builtin_amdgcn_global_load_async_to_lds_b128) && \
    __has_builtin(__builtin_amdgcn_s_wait_asynccnt)
#define USE_ASYNC_LDS 1
#endif
#endif

#ifdef USE_ASYNC_LDS
typedef __attribute__((address_space(1))) v4i as1_v4i;
typedef __attribute__((address_space(3))) v4i as3_v4i;
__device__ __forceinline__ void async_g2l_b128(const void* g, void* l) {
  __builtin_amdgcn_global_load_async_to_lds_b128((as1_v4i*)g, (as3_v4i*)l, 0, 0);
}
__device__ __forceinline__ void async_wait0() { __builtin_amdgcn_s_wait_asynccnt(0); }
#endif

// ======================= WMMA fragment helpers (gfx1250, wave32) =======================
// Layouts per CDNA5 ISA 7.12.2:
//  A (16x32 f16): lane l -> row m=l&15 ; element i -> k = ((i>>3)<<4) + ((l>>4)<<3) + (i&7)
//      => per lane: two contiguous 8-element runs at kbase and kbase+16
//  B (32x16 f16): lane l -> col n=l&15 ; element i -> k = ((i>>3)<<4) + ((l>>4)<<3) + (i&7)
//      => same two 8-element runs when the B operand is stored K-major (B^T[n][k])
//  C/D (16x16 f32): lane l -> col n=l&15 ; slot r -> row m = r + ((l>>4)<<3)

__device__ __forceinline__ v8f wmma32(v16h a, v16h b, v8f c) {
  return __builtin_amdgcn_wmma_f32_16x16x32_f16(false, a, false, b, (short)0, c, false, false);
}

__device__ __forceinline__ v16h fragA_f16(const f16* A, int lda, int row0, int k0, int lane) {
  const f16* Ar = A + (row0 + (lane & 15)) * lda + k0 + ((lane >> 4) << 3);
  v8h lo = *(const v8h*)Ar;
  v8h hi = *(const v8h*)(Ar + 16);
  return __builtin_shufflevector(lo, hi, 0, 1, 2, 3, 4, 5, 6, 7, 8, 9, 10, 11, 12, 13, 14, 15);
}

__device__ __forceinline__ v16h fragA_f32(const float* A, int lda, int row0, int k0, int lane) {
  const float* Ar = A + (row0 + (lane & 15)) * lda + k0 + ((lane >> 4) << 3);
  v8f lo = *(const v8f*)Ar;
  v8f hi = *(const v8f*)(Ar + 16);
  v16h a;
#pragma unroll
  for (int i = 0; i < 8; ++i) { a[i] = (f16)lo[i]; a[i + 8] = (f16)hi[i]; }
  return a;
}

// A fragment with fused rms row-scale rs[] and per-k column weight cs[] (rms(x)*wln)
__device__ __forceinline__ v16h fragA_rms(const float* A, int lda, int row0, int k0,
                                          const float* rs, const float* cs, int lane) {
  int m = row0 + (lane & 15);
  int kbase = k0 + ((lane >> 4) << 3);
  const float* Ar = A + m * lda + kbase;
  v8f lo = *(const v8f*)Ar;
  v8f hi = *(const v8f*)(Ar + 16);
  float r = rs[m];
  v16h a;
#pragma unroll
  for (int i = 0; i < 8; ++i) {
    a[i] = (f16)(lo[i] * r * cs[kbase + i]);
    a[i + 8] = (f16)(hi[i] * r * cs[kbase + 16 + i]);
  }
  return a;
}

// B operand from a K-major (transposed) f16 matrix: B[k][n] = T[n0+n][k]
__device__ __forceinline__ v16h fragBT_f16(const f16* T, int ldk, int n0, int k0, int lane) {
  const f16* Kp = T + (n0 + (lane & 15)) * ldk + k0 + ((lane >> 4) << 3);
  v8h lo = *(const v8h*)Kp;
  v8h hi = *(const v8h*)(Kp + 16);
  return __builtin_shufflevector(lo, hi, 0, 1, 2, 3, 4, 5, 6, 7, 8, 9, 10, 11, 12, 13, 14, 15);
}

// B = (rms(K)*wln)^T with K f32 row-major in LDS (rows are K's rows = our columns)
__device__ __forceinline__ v16h fragBT_rms(const float* Km, int ldk, int n0, int k0,
                                           const float* rs, const float* cs, int lane) {
  int n = n0 + (lane & 15);
  int kbase = k0 + ((lane >> 4) << 3);
  const float* Kp = Km + n * ldk + kbase;
  v8f lo = *(const v8f*)Kp;
  v8f hi = *(const v8f*)(Kp + 16);
  float r = rs[n];
  v16h b;
#pragma unroll
  for (int i = 0; i < 8; ++i) {
    b[i] = (f16)(lo[i] * r * cs[kbase + i]);
    b[i + 8] = (f16)(hi[i] * r * cs[kbase + 16 + i]);
  }
  return b;
}

// ======================= 64x64 matmul helpers (16 waves / 512 threads) =======================
// WT is the 64x64 weight stored K-major (WT[n][k]). Optional DT: f16 K-major copy of D.

__device__ __forceinline__ void mm64_bias(float (*D)[HD], const float (*A)[HD], const f16* WT,
                                          const float* bias, int tid, f16 (*DT)[HD] = nullptr) {
  int lane = tid & 31, w = tid >> 5;
  int row0 = (w & 3) << 4, col0 = (w >> 2) << 4;
  v8f acc = {};
#pragma unroll
  for (int ks = 0; ks < HD; ks += 32) {
    v16h a = fragA_f32(&A[0][0], HD, row0, ks, lane);
    v16h b = fragBT_f16(WT, HD, col0, ks, lane);
    acc = wmma32(a, b, acc);
  }
  int n = col0 + (lane & 15);
  float bb = bias[n];
  int rb = row0 + ((lane >> 4) << 3);
#pragma unroll
  for (int r = 0; r < 8; ++r) {
    float v = acc[r] + bb;
    D[rb + r][n] = v;
    if (DT) DT[n][rb + r] = (f16)v;
  }
}

__device__ __forceinline__ void mm64_scores(float (*S)[HD], const float (*QP)[HD],
                                            const float (*KP)[HD], const float* rsq,
                                            const float* rsk, const float* wln, int tid) {
  int lane = tid & 31, w = tid >> 5;
  int row0 = (w & 3) << 4, col0 = (w >> 2) << 4;
  v8f acc = {};
#pragma unroll
  for (int ks = 0; ks < HD; ks += 32) {
    v16h a = fragA_rms(&QP[0][0], HD, row0, ks, rsq, wln, lane);
    v16h b = fragBT_rms(&KP[0][0], HD, col0, ks, rsk, wln, lane);
    acc = wmma32(a, b, acc);
  }
  int n = col0 + (lane & 15);
  int rb = row0 + ((lane >> 4) << 3);
#pragma unroll
  for (int r = 0; r < 8; ++r) S[rb + r][n] = acc[r] * 0.125f; // 1/sqrt(64)
}

// O = P @ V using a K-major f16 copy of V (VT[d][k])
__device__ __forceinline__ void mm64_pv(float (*O)[HD], const float (*P)[HD],
                                        const f16 (*VT)[HD], int tid) {
  int lane = tid & 31, w = tid >> 5;
  int row0 = (w & 3) << 4, col0 = (w >> 2) << 4;
  v8f acc = {};
#pragma unroll
  for (int ks = 0; ks < HD; ks += 32) {
    v16h a = fragA_f32(&P[0][0], HD, row0, ks, lane);
    v16h b = fragBT_f16(&VT[0][0], HD, col0, ks, lane);
    acc = wmma32(a, b, acc);
  }
  int n = col0 + (lane & 15);
  int rb = row0 + ((lane >> 4) << 3);
#pragma unroll
  for (int r = 0; r < 8; ++r) O[rb + r][n] = acc[r];
}

// ======================= Small (64-row) focus iteration, fully in LDS =======================

struct alignas(32) FocusSmem {
  float curq[HD][HD], kc[HD][HD], vc[HD][HD];
  float qp[HD][HD], kp[HD][HD], vp[HD][HD];
  float sc[HD][HD], attn[HD][HD];
  f16 vpT[HD][HD];
  float rsq[HD], rsk[HD];
  float red[512];
};  // ~142 KB

__device__ void focus_iter(FocusSmem* sm, const f16* WlqT, const f16* WlkT, const f16* WlvT,
                           const float* blq, const float* blk, const float* blv,
                           const float* wln) {
  int tid = threadIdx.x;
  __syncthreads();
  mm64_bias(sm->qp, sm->curq, WlqT, blq, tid);
  mm64_bias(sm->kp, sm->kc, WlkT, blk, tid);
  mm64_bias(sm->vp, sm->vc, WlvT, blv, tid, sm->vpT);
  __syncthreads();
  if (tid < 128) {
    int r = tid & 63;
    const float* src = (tid < 64) ? sm->qp[r] : sm->kp[r];
    float s = 0.f;
    for (int j = 0; j < HD; ++j) s += src[j] * src[j];
    float sc = rsqrtf(s * (1.f / HD) + EPSF);
    if (tid < 64) sm->rsq[r] = sc; else sm->rsk[r] = sc;
  }
  __syncthreads();
  mm64_scores(sm->sc, sm->qp, sm->kp, sm->rsq, sm->rsk, wln, tid);
  __syncthreads();
  if (tid < 64) {
    float m = -3.0e38f;
    for (int j = 0; j < HD; ++j) m = fmaxf(m, sm->sc[tid][j]);
    float sum = 0.f;
    for (int j = 0; j < HD; ++j) { float e = expf(sm->sc[tid][j] - m); sm->sc[tid][j] = e; sum += e; }
    float inv = 1.f / sum;
    for (int j = 0; j < HD; ++j) sm->sc[tid][j] *= inv;
  }
  __syncthreads();
  mm64_pv(sm->attn, sm->sc, sm->vpT, tid);
  __syncthreads();
}

// ======================= Elementwise / setup kernels =======================

__global__ void freqs_kernel(float* f) {
  int i = threadIdx.x;
  if (i < D2) {
    float l21 = log10f(21.0f);
    float mel = powf(10.0f, (float)i * (l21 / 31.0f)) - 1.0f;
    f[i] = (36000.0f / 220.0f) * (200.0f * mel / 1000.0f);
  }
}

// f32 (K x N) -> f16 transposed (N x K)
__global__ void cvtT_kernel(const float* __restrict__ src, f16* __restrict__ dst, int K, int N) {
  int total = K * N;
  for (int idx = blockIdx.x * blockDim.x + threadIdx.x; idx < total;
       idx += gridDim.x * blockDim.x) {
    int k = idx / N, n = idx - k * N;
    dst[(size_t)n * K + k] = (f16)src[idx];
  }
}

__global__ void __launch_bounds__(256) rms2_kernel(const float* __restrict__ X,
                                                   const float* __restrict__ w1,
                                                   const float* __restrict__ w2,
                                                   f16* __restrict__ O1, f16* __restrict__ O2) {
  __shared__ float red[256];
  int row = blockIdx.x, tid = threadIdx.x;
  const float* xr = X + (size_t)row * DIMS;
  float s = 0.f;
  for (int i = tid; i < DIMS; i += 256) { float v = xr[i]; s += v * v; }
  red[tid] = s; __syncthreads();
  for (int st = 128; st > 0; st >>= 1) { if (tid < st) red[tid] += red[tid + st]; __syncthreads(); }
  float sc = rsqrtf(red[0] * (1.f / DIMS) + EPSF);
  for (int i = tid; i < DIMS; i += 256) {
    float v = xr[i] * sc;
    O1[(size_t)row * DIMS + i] = (f16)(v * w1[i]);
    if (O2) O2[(size_t)row * DIMS + i] = (f16)(v * w2[i]);
  }
}

// RoPE in place over full rows (t = row % CTXN), pairs (2i,2i+1) within each head
__global__ void rope_rows_kernel(float* __restrict__ Q, const float* __restrict__ freqs) {
  int total = NROWS * (DIMS / 2);
  for (int p = blockIdx.x * blockDim.x + threadIdx.x; p < total; p += gridDim.x * blockDim.x) {
    int row = p >> 9, pp = p & 511;
    int i = pp & 31, col = ((pp >> 5) << 6) + (i << 1);
    float ang = (float)(row & (CTXN - 1)) * freqs[i];
    float c = cosf(ang), s = sinf(ang);
    float* r = Q + (size_t)row * DIMS + col;
    float x1 = r[0], x2 = r[1];
    r[0] = x1 * c - x2 * s;
    r[1] = x1 * s + x2 * c;
  }
}

// split kv (rows x 2048) into roped K (rows x 1024) and plain V (rows x 1024)
__global__ void split_kv_kernel(const float* __restrict__ KV, float* __restrict__ Kc,
                                float* __restrict__ Vc, const float* __restrict__ freqs) {
  int total = NROWS * (DIMS / 2);
  for (int p = blockIdx.x * blockDim.x + threadIdx.x; p < total; p += gridDim.x * blockDim.x) {
    int row = p >> 9, pp = p & 511;
    int i = pp & 31, col = ((pp >> 5) << 6) + (i << 1);
    float ang = (float)(row & (CTXN - 1)) * freqs[i];
    float c = cosf(ang), s = sinf(ang);
    const float* kr = KV + (size_t)row * (2 * DIMS) + col;
    float x1 = kr[0], x2 = kr[1];
    Kc[(size_t)row * DIMS + col] = x1 * c - x2 * s;
    Kc[(size_t)row * DIMS + col + 1] = x1 * s + x2 * c;
    const float* vr = KV + (size_t)row * (2 * DIMS) + DIMS + col;
    Vc[(size_t)row * DIMS + col] = vr[0];
    Vc[(size_t)row * DIMS + col + 1] = vr[1];
  }
}

__global__ void addinto_kernel(float* __restrict__ dst, const float* __restrict__ src, int n,
                               const int* gate) {
  if (gate && *gate) return;
  for (int i = blockIdx.x * blockDim.x + threadIdx.x; i < n; i += gridDim.x * blockDim.x)
    dst[i] += src[i];
}

__global__ void __launch_bounds__(256) diffsum_kernel(const float* __restrict__ a,
                                                      const float* __restrict__ b, int n,
                                                      float* out) {
  __shared__ float red[256];
  int tid = threadIdx.x;
  float s = 0.f;
  for (int i = blockIdx.x * 256 + tid; i < n; i += gridDim.x * 256) s += fabsf(a[i] - b[i]);
  red[tid] = s; __syncthreads();
  for (int st = 128; st > 0; st >>= 1) { if (tid < st) red[tid] += red[tid + st]; __syncthreads(); }
  if (tid == 0) atomicAdd(out, red[0]);
}

__global__ void flag_kernel(const float* dsum, int* done, int n, float inv) {
  int i = blockIdx.x * blockDim.x + threadIdx.x;
  if (i < n) done[i] = (dsum[i] * inv < 0.5f) ? 1 : 0;
}

// ======================= Big GEMM (64x64 tiles, 8 waves, async LDS staging) =======================
// BT is the weight stored K-major: BT[n][k], n in [0,N), k in [0,K).

__global__ void __launch_bounds__(256) gemm_kernel(
    const f16* __restrict__ A16, const float* __restrict__ A32a, const float* __restrict__ A32b,
    const int* __restrict__ flag, const f16* __restrict__ BT, const float* __restrict__ bias,
    float* __restrict__ C, int M, int N, int K) {
  __shared__ alignas(32) f16 As[64][32];
  __shared__ alignas(32) f16 BsT[64][32];
  int tid = threadIdx.x, lane = tid & 31, w = tid >> 5;
  int row0 = blockIdx.y * 64, col0 = blockIdx.x * 64;
  const float* A32 = (A16 == nullptr) ? ((flag && *flag) ? A32a : A32b) : nullptr;
  int tm = (w & 3) << 4;
  int tn0 = (w >> 2) << 5;
  v8f acc0 = {}, acc1 = {};
  for (int k0 = 0; k0 < K; k0 += 32) {
    __syncthreads();
    if (k0 + 32 < K) {  // prefetch next K panels
      if (A16) __builtin_prefetch(A16 + (size_t)(row0 + (tid >> 2)) * K + k0 + 32, 0, 3);
      __builtin_prefetch(BT + (size_t)(col0 + (tid >> 2)) * K + k0 + 32, 0, 3);
    }
#ifdef USE_ASYNC_LDS
    if (A16) {
      int r = tid >> 2, c8 = (tid & 3) << 3;
      async_g2l_b128(A16 + (size_t)(row0 + r) * K + k0 + c8, &As[r][c8]);
    } else {
      for (int e = tid; e < 64 * 32; e += 256) {
        int r = e >> 5, c = e & 31;
        As[r][c] = (f16)A32[(size_t)(row0 + r) * K + k0 + c];
      }
    }
    {
      int c = tid >> 2, r8 = (tid & 3) << 3;
      async_g2l_b128(BT + (size_t)(col0 + c) * K + k0 + r8, &BsT[c][r8]);
    }
    async_wait0();
#else
    for (int e = tid; e < 64 * 32; e += 256) {
      int r = e >> 5, c = e & 31;
      As[r][c] = A16 ? A16[(size_t)(row0 + r) * K + k0 + c]
                     : (f16)A32[(size_t)(row0 + r) * K + k0 + c];
    }
    for (int e = tid; e < 64 * 32; e += 256) {
      int c = e >> 5, r = e & 31;
      BsT[c][r] = BT[(size_t)(col0 + c) * K + k0 + r];
    }
#endif
    __syncthreads();
    v16h a = fragA_f16(&As[0][0], 32, tm, 0, lane);
    v16h b0 = fragBT_f16(&BsT[0][0], 32, tn0, 0, lane);
    v16h b1 = fragBT_f16(&BsT[0][0], 32, tn0 + 16, 0, lane);
    acc0 = wmma32(a, b0, acc0);
    acc1 = wmma32(a, b1, acc1);
  }
  int n0 = col0 + tn0 + (lane & 15);
  int n1 = n0 + 16;
  int rb = row0 + tm + ((lane >> 4) << 3);
  float bb0 = bias[n0], bb1 = bias[n1];
#pragma unroll
  for (int r = 0; r < 8; ++r) {
    C[(size_t)(rb + r) * N + n0] = acc0[r] + bb0;
    C[(size_t)(rb + r) * N + n1] = acc1[r] + bb1;
  }
}

// ======================= Slide-stage attention (one block per window/batch/head) =======================

__global__ void __launch_bounds__(512) slide_attn_ab(
    const float* __restrict__ qfull, const float* __restrict__ kvfull,
    const f16* WlqT, const f16* WlkT, const f16* WlvT,
    const float* blq, const float* blk, const float* blv, const float* wln,
    const float* __restrict__ freqs,
    float* __restrict__ xw, float* __restrict__ scurq, float* __restrict__ skc,
    float* __restrict__ svc, float* __restrict__ diffsum) {
  __shared__ FocusSmem sm;
  int h = blockIdx.x, b = blockIdx.y, win = blockIdx.z;
  int tid = threadIdx.x;
  int qrow0 = b * CTXN + win * HD;
  int kstart = win * HD - 4; if (kstart < 0) kstart = 0;
  int krow0 = b * CTXN + kstart;
  // gather + RoPE (window-local t) into LDS
  for (int e = tid; e < HD * D2; e += 512) {
    int t = e >> 5, i = e & 31;
    float ang = (float)t * freqs[i];
    float c = cosf(ang), s = sinf(ang);
    int c0 = h * HD + 2 * i;
    {
      const float* qr = qfull + (size_t)(qrow0 + t) * DIMS + c0;
      float x1 = qr[0], x2 = qr[1];
      sm.curq[t][2 * i] = x1 * c - x2 * s;
      sm.curq[t][2 * i + 1] = x1 * s + x2 * c;
    }
    {
      const float* kr = kvfull + (size_t)(krow0 + t) * (2 * DIMS) + c0;
      float x1 = kr[0], x2 = kr[1];
      sm.kc[t][2 * i] = x1 * c - x2 * s;
      sm.kc[t][2 * i + 1] = x1 * s + x2 * c;
    }
    {
      const float* vr = kvfull + (size_t)(krow0 + t) * (2 * DIMS) + DIMS + c0;
      sm.vc[t][2 * i] = vr[0];
      sm.vc[t][2 * i + 1] = vr[1];
    }
  }
  // iteration 0
  focus_iter(&sm, WlqT, WlkT, WlvT, blq, blk, blv, wln);
  float pv[8];
  {
    int j = 0;
    for (int e = tid; e < HD * HD; e += 512, ++j) {
      int t = e >> 6, d = e & 63;
      float a = sm.attn[t][d];
      pv[j] = a;
      sm.curq[t][d] += a;
      sm.kc[t][d] = sm.kp[t][d];
      sm.vc[t][d] = sm.vp[t][d];
    }
  }
  // iteration 1 (focus_iter begins with a barrier)
  focus_iter(&sm, WlqT, WlkT, WlvT, blq, blk, blv, wln);
  // |attn1 - attn0| partial sum
  float loc = 0.f;
  {
    int j = 0;
    for (int e = tid; e < HD * HD; e += 512, ++j) {
      int t = e >> 6, d = e & 63;
      loc += fabsf(sm.attn[t][d] - pv[j]);
    }
  }
  sm.red[tid] = loc; __syncthreads();
  for (int st = 256; st > 0; st >>= 1) { if (tid < st) sm.red[tid] += sm.red[tid + st]; __syncthreads(); }
  if (tid == 0) atomicAdd(&diffsum[win], sm.red[0]);
  // write attn1 (output-if-done) and state for a possible iteration 2
  for (int e = tid; e < HD * HD; e += 512) {
    int t = e >> 6, d = e & 63;
    size_t idx = (size_t)(qrow0 + t) * DIMS + h * HD + d;
    float a = sm.attn[t][d];
    xw[idx] = a;
    scurq[idx] = sm.curq[t][d] + a;
    skc[idx] = sm.kp[t][d];
    svc[idx] = sm.vp[t][d];
  }
}

__global__ void __launch_bounds__(512) slide_attn_c(
    const float* __restrict__ scurq, const float* __restrict__ skc, const float* __restrict__ svc,
    const int* __restrict__ done,
    const f16* WlqT, const f16* WlkT, const f16* WlvT,
    const float* blq, const float* blk, const float* blv, const float* wln,
    float* __restrict__ xw) {
  int h = blockIdx.x, b = blockIdx.y, win = blockIdx.z;
  if (done[win]) return;
  __shared__ FocusSmem sm;
  int tid = threadIdx.x;
  int row0 = b * CTXN + win * HD;
  for (int e = tid; e < HD * HD; e += 512) {
    int t = e >> 6, d = e & 63;
    size_t idx = (size_t)(row0 + t) * DIMS + h * HD + d;
    sm.curq[t][d] = scurq[idx];
    sm.kc[t][d] = skc[idx];
    sm.vc[t][d] = svc[idx];
  }
  focus_iter(&sm, WlqT, WlkT, WlvT, blq, blk, blv, wln);
  for (int e = tid; e < HD * HD; e += 512) {
    int t = e >> 6, d = e & 63;
    xw[(size_t)(row0 + t) * DIMS + h * HD + d] = sm.attn[t][d];
  }
}

// ======================= Final-stage projection + flash attention =======================

struct alignas(32) ProjSmem {
  float aq[HD][HD], ak[HD][HD], av[HD][HD];
  float qp[HD][HD], kp[HD][HD], vp[HD][HD];
  float rsq[HD], rsk[HD];
};  // ~96.5 KB

__global__ void __launch_bounds__(512) proj_kernel(
    const float* __restrict__ curq, const float* __restrict__ kc, const float* __restrict__ vc,
    const f16* WlqT, const f16* WlkT, const f16* WlvT,
    const float* blq, const float* blk, const float* blv, const float* wln,
    f16* __restrict__ qpn, f16* __restrict__ kpn, f16* __restrict__ vph,
    float* __restrict__ kc_out, float* __restrict__ vc_out, const int* gate) {
  if (gate && *gate) return;
  __shared__ ProjSmem sm;
  int h = blockIdx.x, rt = blockIdx.y, tid = threadIdx.x;
  int row0 = rt * 64, c0 = h * HD;
  for (int e = tid; e < HD * HD; e += 512) {
    int t = e >> 6, d = e & 63;
    size_t idx = (size_t)(row0 + t) * DIMS + c0 + d;
    sm.aq[t][d] = curq[idx];
    sm.ak[t][d] = kc[idx];
    sm.av[t][d] = vc[idx];
  }
  __syncthreads();
  mm64_bias(sm.qp, sm.aq, WlqT, blq, tid);
  mm64_bias(sm.kp, sm.ak, WlkT, blk, tid);
  mm64_bias(sm.vp, sm.av, WlvT, blv, tid);
  __syncthreads();
  if (tid < 128) {
    int r = tid & 63;
    const float* src = (tid < 64) ? sm.qp[r] : sm.kp[r];
    float s = 0.f;
    for (int j = 0; j < HD; ++j) s += src[j] * src[j];
    float sc = rsqrtf(s * (1.f / HD) + EPSF);
    if (tid < 64) sm.rsq[r] = sc; else sm.rsk[r] = sc;
  }
  __syncthreads();
  for (int e = tid; e < HD * HD; e += 512) {
    int t = e >> 6, d = e & 63;
    size_t idx = (size_t)(row0 + t) * DIMS + c0 + d;
    qpn[idx] = (f16)(sm.qp[t][d] * sm.rsq[t] * wln[d]);
    kpn[idx] = (f16)(sm.kp[t][d] * sm.rsk[t] * wln[d]);
    vph[idx] = (f16)sm.vp[t][d];
    if (kc_out) { kc_out[idx] = sm.kp[t][d]; vc_out[idx] = sm.vp[t][d]; }
  }
}

struct alignas(32) FlashSmem {
  f16 qn[HD][HD], kn[HD][HD];
  f16 vtT[HD][HD];  // K-major copy of the V tile: vtT[d][keyrow]
  float s[HD][HD], acc[HD][HD];
  float rowm[HD], rowl[HD];
};  // ~56.5 KB

__global__ void __launch_bounds__(512) flash_kernel(
    const f16* __restrict__ qpn, const f16* __restrict__ kpn, const f16* __restrict__ vph,
    float* __restrict__ attn, const int* gate) {
  if (gate && *gate) return;
  __shared__ FlashSmem sm;
  int h = blockIdx.x, qt = blockIdx.y, b = blockIdx.z;
  int tid = threadIdx.x, lane = tid & 31, w = tid >> 5;
  int tm = (w & 3) << 4, tn = (w >> 2) << 4;
  int qrow0 = b * CTXN + qt * 64, c0 = h * HD;
  for (int e = tid; e < HD * HD; e += 512) {
    int t = e >> 6, d = e & 63;
    sm.qn[t][d] = qpn[(size_t)(qrow0 + t) * DIMS + c0 + d];
    sm.acc[t][d] = 0.f;
  }
  if (tid < 64) { sm.rowm[tid] = -3.0e38f; sm.rowl[tid] = 0.f; }
  for (int kt = 0; kt < CTXN / 64; ++kt) {
    int krow0 = b * CTXN + kt * 64;
    __syncthreads();
    for (int e = tid; e < HD * HD; e += 512) {
      int t = e >> 6, d = e & 63;
      sm.kn[t][d] = kpn[(size_t)(krow0 + t) * DIMS + c0 + d];
      sm.vtT[d][t] = vph[(size_t)(krow0 + t) * DIMS + c0 + d];
    }
    __syncthreads();
    // scores tile = qn @ kn^T * 0.125
    {
      v8f acc = {};
#pragma unroll
      for (int ks = 0; ks < HD; ks += 32) {
        v16h a = fragA_f16(&sm.qn[0][0], HD, tm, ks, lane);
        v16h bt = fragBT_f16(&sm.kn[0][0], HD, tn, ks, lane);
        acc = wmma32(a, bt, acc);
      }
      int n = tn + (lane & 15);
      int rb = tm + ((lane >> 4) << 3);
#pragma unroll
      for (int r = 0; r < 8; ++r) sm.s[rb + r][n] = acc[r] * 0.125f;
    }
    __syncthreads();
    if (tid < 64) {
      int t = tid;
      float m = sm.rowm[t];
      float mk = -3.0e38f;
      for (int j = 0; j < HD; ++j) mk = fmaxf(mk, sm.s[t][j]);
      float nm = fmaxf(m, mk);
      float corr = expf(m - nm);
      float ssum = 0.f;
      for (int j = 0; j < HD; ++j) { float p = expf(sm.s[t][j] - nm); sm.s[t][j] = p; ssum += p; }
      sm.rowl[t] = sm.rowl[t] * corr + ssum;
      sm.rowm[t] = nm;
      for (int j = 0; j < HD; ++j) sm.acc[t][j] *= corr;
    }
    __syncthreads();
    // acc += p @ v
    {
      int n = tn + (lane & 15);
      int rb = tm + ((lane >> 4) << 3);
      v8f c;
#pragma unroll
      for (int r = 0; r < 8; ++r) c[r] = sm.acc[rb + r][n];
#pragma unroll
      for (int ks = 0; ks < HD; ks += 32) {
        v16h a = fragA_f32(&sm.s[0][0], HD, tm, ks, lane);
        v16h bv = fragBT_f16(&sm.vtT[0][0], HD, tn, ks, lane);
        c = wmma32(a, bv, c);
      }
#pragma unroll
      for (int r = 0; r < 8; ++r) sm.acc[rb + r][n] = c[r];
    }
  }
  __syncthreads();
  for (int e = tid; e < HD * HD; e += 512) {
    int t = e >> 6, d = e & 63;
    attn[(size_t)(qrow0 + t) * DIMS + c0 + d] = sm.acc[t][d] / sm.rowl[t];
  }
}

// ======================= Host-side orchestration =======================

static void launch_slide(const float* src, float* dst,
                         const float* wq_norm, const float* wkv_norm,
                         const f16* WqT16, const float* bq, const f16* WkvT16, const float* bkv,
                         const f16* WlqT16, const f16* WlkT16, const f16* WlvT16,
                         const float* blq, const float* blk, const float* blv, const float* wln,
                         const float* freqs,
                         f16* xqn, f16* xkvn, float* qfull, float* kvfull,
                         float* scurq, float* skc, float* svc,
                         float* diffsum, int* done, hipStream_t stream) {
  rms2_kernel<<<NROWS, 256, 0, stream>>>(src, wq_norm, wkv_norm, xqn, xkvn);
  gemm_kernel<<<dim3(DIMS / 64, NROWS / 64), 256, 0, stream>>>(
      xqn, nullptr, nullptr, nullptr, WqT16, bq, qfull, NROWS, DIMS, DIMS);
  gemm_kernel<<<dim3(2 * DIMS / 64, NROWS / 64), 256, 0, stream>>>(
      xkvn, nullptr, nullptr, nullptr, WkvT16, bkv, kvfull, NROWS, 2 * DIMS, DIMS);
  (void)hipMemsetAsync(diffsum, 0, NWIN * sizeof(float), stream);
  slide_attn_ab<<<dim3(HEADS, NB, NWIN), 512, 0, stream>>>(
      qfull, kvfull, WlqT16, WlkT16, WlvT16, blq, blk, blv, wln, freqs,
      dst, scurq, skc, svc, diffsum);
  flag_kernel<<<1, NWIN, 0, stream>>>(diffsum, done, NWIN, 1.0f / (NB * HEADS * HD * HD));
  slide_attn_c<<<dim3(HEADS, NB, NWIN), 512, 0, stream>>>(
      scurq, skc, svc, done, WlqT16, WlkT16, WlvT16, blq, blk, blv, wln, dst);
}

extern "C" void kernel_launch(void* const* d_in, const int* in_sizes, int n_in,
                              void* d_out, int out_size, void* d_ws, size_t ws_size,
                              hipStream_t stream) {
  const float* x       = (const float*)d_in[0];
  const float* xa      = (const float*)d_in[1];
  const float* wq_norm = (const float*)d_in[2];
  const float* Wq      = (const float*)d_in[3];
  const float* bq      = (const float*)d_in[4];
  const float* wkv_nm  = (const float*)d_in[5];
  const float* Wkv     = (const float*)d_in[6];
  const float* bkv     = (const float*)d_in[7];
  const float* wln     = (const float*)d_in[8];
  const float* Wlq     = (const float*)d_in[9];
  const float* blq     = (const float*)d_in[10];
  const float* Wlk     = (const float*)d_in[11];
  const float* blk     = (const float*)d_in[12];
  const float* Wlv     = (const float*)d_in[13];
  const float* blv     = (const float*)d_in[14];
  const float* Wo      = (const float*)d_in[15];
  const float* bo      = (const float*)d_in[16];
  float* out = (float*)d_out;

  char* ws = (char*)d_ws;
  size_t off = 0;
  auto take = [&](size_t bytes) -> void* {
    void* p = ws + off;
    off = (off + bytes + 255) & ~(size_t)255;
    return p;
  };
  float* qfull  = (float*)take((size_t)NELEM * 4);
  float* kvfull = (float*)take((size_t)2 * NELEM * 4);
  float* xw     = (float*)take((size_t)NELEM * 4);
  float* xaw    = (float*)take((size_t)NELEM * 4);
  float* bufA   = (float*)take((size_t)NELEM * 4);  // slide scurq / final kcA, kp1
  float* bufB   = (float*)take((size_t)NELEM * 4);  // slide skc   / final vcA, vp1
  float* bufC   = (float*)take((size_t)NELEM * 4);  // slide svc
  float* attnA  = (float*)take((size_t)NELEM * 4);
  float* attnB  = (float*)take((size_t)NELEM * 4);
  f16* xqn  = (f16*)take((size_t)NELEM * 2);  // also qpn in final stage
  f16* xkvn = (f16*)take((size_t)NELEM * 2);  // also kpn in final stage
  f16* vph  = (f16*)take((size_t)NELEM * 2);
  f16* WqT16  = (f16*)take((size_t)DIMS * DIMS * 2);
  f16* WkvT16 = (f16*)take((size_t)DIMS * 2 * DIMS * 2);
  f16* WoT16  = (f16*)take((size_t)DIMS * DIMS * 2);
  f16* WlqT16 = (f16*)take((size_t)HD * HD * 2);
  f16* WlkT16 = (f16*)take((size_t)HD * HD * 2);
  f16* WlvT16 = (f16*)take((size_t)HD * HD * 2);
  float* freqs   = (float*)take(D2 * 4);
  float* diffsum = (float*)take(NWIN * 4);
  int* done      = (int*)take(NWIN * 4);
  float* dsum    = (float*)take(4);
  int* doneF     = (int*)take(4);
  (void)in_sizes; (void)n_in; (void)out_size; (void)ws_size;

  // kvfull is dead after split_kv in the final stage: reuse it for kp0/vp0
  float* kcB = kvfull;
  float* vcB = kvfull + NELEM;

  // ---- setup: freqs + transposed f16 weights (K-major B operands)
  freqs_kernel<<<1, 32, 0, stream>>>(freqs);
  cvtT_kernel<<<2048, 256, 0, stream>>>(Wq, WqT16, DIMS, DIMS);
  cvtT_kernel<<<2048, 256, 0, stream>>>(Wkv, WkvT16, DIMS, 2 * DIMS);
  cvtT_kernel<<<2048, 256, 0, stream>>>(Wo, WoT16, DIMS, DIMS);
  cvtT_kernel<<<16, 256, 0, stream>>>(Wlq, WlqT16, HD, HD);
  cvtT_kernel<<<16, 256, 0, stream>>>(Wlk, WlkT16, HD, HD);
  cvtT_kernel<<<16, 256, 0, stream>>>(Wlv, WlvT16, HD, HD);

  // ---- sliding-window focus on x and xa
  launch_slide(x, xw, wq_norm, wkv_nm, WqT16, bq, WkvT16, bkv, WlqT16, WlkT16, WlvT16,
               blq, blk, blv, wln, freqs, xqn, xkvn, qfull, kvfull,
               bufA, bufB, bufC, diffsum, done, stream);
  launch_slide(xa, xaw, wq_norm, wkv_nm, WqT16, bq, WkvT16, bkv, WlqT16, WlkT16, WlvT16,
               blq, blk, blv, wln, freqs, xqn, xkvn, qfull, kvfull,
               bufA, bufB, bufC, diffsum, done, stream);

  // ---- final focus(xw, xaw)
  rms2_kernel<<<NROWS, 256, 0, stream>>>(xw, wq_norm, nullptr, xqn, nullptr);
  rms2_kernel<<<NROWS, 256, 0, stream>>>(xaw, wkv_nm, nullptr, xkvn, nullptr);
  gemm_kernel<<<dim3(DIMS / 64, NROWS / 64), 256, 0, stream>>>(
      xqn, nullptr, nullptr, nullptr, WqT16, bq, qfull, NROWS, DIMS, DIMS);
  gemm_kernel<<<dim3(2 * DIMS / 64, NROWS / 64), 256, 0, stream>>>(
      xkvn, nullptr, nullptr, nullptr, WkvT16, bkv, kvfull, NROWS, 2 * DIMS, DIMS);
  rope_rows_kernel<<<2048, 256, 0, stream>>>(qfull, freqs);             // qfull becomes curq
  split_kv_kernel<<<2048, 256, 0, stream>>>(kvfull, bufA, bufB, freqs); // kcA=bufA, vcA=bufB

  f16* qpn = xqn; f16* kpn = xkvn;
  // iteration 0
  proj_kernel<<<dim3(HEADS, NROWS / 64), 512, 0, stream>>>(
      qfull, bufA, bufB, WlqT16, WlkT16, WlvT16, blq, blk, blv, wln,
      qpn, kpn, vph, kcB, vcB, nullptr);
  flash_kernel<<<dim3(HEADS, CTXN / 64, NB), 512, 0, stream>>>(qpn, kpn, vph, attnA, nullptr);
  addinto_kernel<<<4096, 256, 0, stream>>>(qfull, attnA, NELEM, nullptr);
  // iteration 1 (kp1/vp1 saved to bufA/bufB for a possible iteration 2)
  proj_kernel<<<dim3(HEADS, NROWS / 64), 512, 0, stream>>>(
      qfull, kcB, vcB, WlqT16, WlkT16, WlvT16, blq, blk, blv, wln,
      qpn, kpn, vph, bufA, bufB, nullptr);
  flash_kernel<<<dim3(HEADS, CTXN / 64, NB), 512, 0, stream>>>(qpn, kpn, vph, attnB, nullptr);
  (void)hipMemsetAsync(dsum, 0, sizeof(float), stream);
  diffsum_kernel<<<256, 256, 0, stream>>>(attnB, attnA, NELEM, dsum);
  flag_kernel<<<1, 1, 0, stream>>>(dsum, doneF, 1, 1.0f / NELEM);
  // iteration 2 (runs only if not done)
  addinto_kernel<<<4096, 256, 0, stream>>>(qfull, attnB, NELEM, doneF);
  proj_kernel<<<dim3(HEADS, NROWS / 64), 512, 0, stream>>>(
      qfull, bufA, bufB, WlqT16, WlkT16, WlvT16, blq, blk, blv, wln,
      qpn, kpn, vph, nullptr, nullptr, doneF);
  flash_kernel<<<dim3(HEADS, CTXN / 64, NB), 512, 0, stream>>>(qpn, kpn, vph, attnA, doneF);
  // epilogue: out = (done ? attn1 : attn2) @ Wo + bo
  gemm_kernel<<<dim3(DIMS / 64, NROWS / 64), 256, 0, stream>>>(
      nullptr, attnB, attnA, doneF, WoT16, bo, out, NROWS, DIMS, DIMS);
}
